// EdgeAwareTransformerBackbone_17265768530359
// MI455X (gfx1250) — matrix operand order repaired and checked by reading, not compiled, not run on hardware
//
#include <hip/hip_runtime.h>
#include <hip/hip_bf16.h>

// ---------------- problem constants ----------------
#define kN    50000     // nodes
#define kE    100000    // edges
#define kHID  128       // hidden
#define kH    4         // heads
#define kC    128       // channels per head
#define kHC   512       // H*C
#define kL    3         // layers
#define kSCALE 0.08838834764831845f   // 1/sqrt(128)

// ---------------- WMMA types -----------------------
typedef __attribute__((ext_vector_type(16))) __bf16 v16bf;
typedef __attribute__((ext_vector_type(8)))  float  v8f;

__device__ __forceinline__ v8f wmma_bf16(v16bf a, v16bf b, v8f c) {
    // v_wmma_f32_16x16x32_bf16  D = A(16x32) x B(32x16) + C
    return __builtin_amdgcn_wmma_f32_16x16x32_bf16(
        false, a, false, b, (short)0, c, false, false);
}

// A fragment from bf16 LDS tile: 16x32.
// lane 0-15 : row=lane,    K = k0 + {0..15}
// lane 16-31: row=lane-16, K = k0 + {16..31}
__device__ __forceinline__ void load_a_frag(v16bf& a, const __bf16* src, int ld,
                                            int k0, int lane) {
    int row = lane & 15;
    int kb  = (lane >> 4) << 4;             // 0 or 16
    const __bf16* p = src + row * ld + k0 + kb;
#pragma unroll
    for (int j = 0; j < 8; ++j) a[j]     = p[j];
#pragma unroll
    for (int j = 0; j < 8; ++j) a[8 + j] = p[8 + j];
}

// B fragment from pre-packed buffer: fragment f = 32 lanes x 16 bf16,
// lane's 16 values contiguous -> one 32B coalesced read per lane.
__device__ __forceinline__ v16bf load_b_packed(const v16bf* frags,
                                               int fragIdx, int lane) {
    return frags[(size_t)fragIdx * 32 + lane];
}

// monotone float->uint key so unsigned atomicMax == float max
__device__ __forceinline__ unsigned fkey(float f) {
    unsigned u = __float_as_uint(f);
    return (u & 0x80000000u) ? ~u : (u | 0x80000000u);
}
__device__ __forceinline__ float fkey_dec(unsigned k) {
    unsigned u = (k & 0x80000000u) ? (k ^ 0x80000000u) : ~k;
    return __uint_as_float(u);
}

// ---------------- kernels --------------------------

__global__ void zero_kernel(float* __restrict__ p, int n) {
    int i = blockIdx.x * blockDim.x + threadIdx.x;
    if (i < n) p[i] = 0.0f;
}

// Re-pack W[K,Nn] (fp32 row-major) into bf16 B-fragment order:
// frag (nt, ks): 32 lanes x 16 values; lane half h holds k = 16h + j (contiguous, same order as A):
//   col = lane&15, kb = (lane>>4)*16, value j -> W[(ks*32+kb+j)*Nn + nt*16+col]
__global__ void pack_w_kernel(const float* __restrict__ W, int K, int Nn,
                              __bf16* __restrict__ out) {
    int o = blockIdx.x * blockDim.x + threadIdx.x;
    int total = K * Nn;
    if (o >= total) return;
    int ksteps = K >> 5;
    int fragid = o >> 9;                 // / 512
    int within = o & 511;
    int lane = within >> 4, j = within & 15;
    int nt = fragid / ksteps, ks = fragid - nt * ksteps;
    int col = lane & 15, kb = (lane >> 4) << 4;
    int kk = (ks << 5) + kb + j;
    out[o] = (__bf16)W[(size_t)kk * Nn + nt * 16 + col];
}

// x[N,128] @ {Wq,Wk,Wv}[128,512] + bias -> q,k,v[N,512]
// grid (N/16, 12), block 256 (8 waves). wave -> global ntile g in [0,96):
// g/32 selects matrix, g%32 selects 16-col tile.
__global__ __launch_bounds__(256) void qkv_gemm_kernel(
    const float* __restrict__ x,
    const v16bf* __restrict__ WqP, const v16bf* __restrict__ WkP,
    const v16bf* __restrict__ WvP,
    const float* __restrict__ bq, const float* __restrict__ bk,
    const float* __restrict__ bv,
    float* __restrict__ q, float* __restrict__ k, float* __restrict__ v)
{
    __shared__ __bf16 xsb[16 * kHID];    // 4 KB, bf16 staged once
    int tid = threadIdx.x;
    int m0  = blockIdx.x * 16;
    for (int i = tid; i < 16 * kHID; i += 256)
        xsb[i] = (__bf16)x[(size_t)(m0 + (i >> 7)) * kHID + (i & 127)];
    __syncthreads();

    int wv_  = tid >> 5;
    int lane = tid & 31;
    int g    = blockIdx.y * 8 + wv_;         // 0..95
    int mat  = g >> 5;
    int nt   = g & 31;
    int n0   = nt * 16;

    const v16bf* WP; const float* bias; float* out;
    if (mat == 0)      { WP = WqP; bias = bq; out = q; }
    else if (mat == 1) { WP = WkP; bias = bk; out = k; }
    else               { WP = WvP; bias = bv; out = v; }

    v8f acc = {};
#pragma unroll
    for (int ks = 0; ks < 4; ++ks) {         // K = 128
        v16bf a;
        load_a_frag(a, xsb, kHID, ks << 5, lane);
        v16bf b = load_b_packed(WP, nt * 4 + ks, lane);
        acc = wmma_bf16(a, b, acc);
    }
    int col = lane & 15;
    int mb  = (lane >> 4) << 3;              // 0 or 8
    float bsc = bias[n0 + col];
#pragma unroll
    for (int r = 0; r < 8; ++r)
        out[(size_t)(m0 + mb + r) * kHC + n0 + col] = acc[r] + bsc;
}

// one wave per edge: alpha logits + segment max (monotone-key atomicMax)
__global__ __launch_bounds__(256) void edge_alpha_kernel(
    const int* __restrict__ ei, const float* __restrict__ ea,
    const float* __restrict__ q, const float* __restrict__ k,
    const float* __restrict__ We,
    float* __restrict__ alpha, unsigned* __restrict__ maxkey)
{
    int e    = (blockIdx.x * blockDim.x + threadIdx.x) >> 5;
    int lane = threadIdx.x & 31;
    if (e >= kE) return;
    int src = ei[e], dst = ei[kE + e];
    float eav = ea[e];
    const float* qp = q + (size_t)dst * kHC;
    const float* kp = k + (size_t)src * kHC;
    int base = lane * 16;
    float p = 0.0f;
#pragma unroll
    for (int i = 0; i < 16; ++i) {
        int c = base + i;
        p += qp[c] * (kp[c] + eav * We[c]);
    }
    // reduce within 8-lane group (= one head: 128 channels / 16 per lane)
    p += __shfl_xor(p, 1);
    p += __shfl_xor(p, 2);
    p += __shfl_xor(p, 4);
    if ((lane & 7) == 0) {
        int h = lane >> 3;
        float a = p * kSCALE;
        alpha[(size_t)e * kH + h] = a;
        atomicMax(&maxkey[(size_t)dst * kH + h], fkey(a));
    }
}

// one thread per (edge, head): p = exp(alpha - max), denom += p
__global__ void edge_exp_kernel(
    const int* __restrict__ ei, const float* __restrict__ alpha,
    const unsigned* __restrict__ maxkey,
    float* __restrict__ pbuf, float* __restrict__ denom)
{
    int i = blockIdx.x * blockDim.x + threadIdx.x;
    if (i >= kE * kH) return;
    int e = i >> 2, h = i & 3;
    int dst = ei[kE + e];
    float p = __expf(alpha[i] - fkey_dec(maxkey[(size_t)dst * kH + h]));
    pbuf[i] = p;
    atomicAdd(&denom[(size_t)dst * kH + h], p);
}

// one wave per edge: agg[dst] += (v[src] + e) * a
__global__ __launch_bounds__(256) void edge_scatter_kernel(
    const int* __restrict__ ei, const float* __restrict__ ea,
    const float* __restrict__ v, const float* __restrict__ We,
    const float* __restrict__ pbuf, const float* __restrict__ denom,
    float* __restrict__ agg)
{
    int e    = (blockIdx.x * blockDim.x + threadIdx.x) >> 5;
    int lane = threadIdx.x & 31;
    if (e >= kE) return;
    int src = ei[e], dst = ei[kE + e];
    float eav = ea[e];
    int h = lane >> 3;
    float a = pbuf[(size_t)e * kH + h] /
              (denom[(size_t)dst * kH + h] + 1e-16f);
    const float* vp = v + (size_t)src * kHC;
    float* ap = agg + (size_t)dst * kHC;
    int base = lane * 16;
#pragma unroll
    for (int i = 0; i < 16; ++i) {
        int c = base + i;
        atomicAdd(&ap[c], (vp[c] + eav * We[c]) * a);
    }
}

// fused: conv = agg + x@Wskip + bskip ;  h = elu(conv@Wproj + bproj)
//        y = x + h ; out = LayerNorm(y)*gamma + beta
// grid N/16, block 256 (8 waves). conv tile lives in LDS only (as bf16,
// since it is consumed solely as a WMMA A-operand).
__global__ __launch_bounds__(256) void conv_proj_ln_kernel(
    const float* __restrict__ x, const float* __restrict__ agg,
    const v16bf* __restrict__ WskipP, const float* __restrict__ bskip,
    const v16bf* __restrict__ WprojP, const float* __restrict__ bproj,
    const float* __restrict__ gamma, const float* __restrict__ beta,
    float* __restrict__ out)
{
    __shared__ float  xs[16 * kHID];      //  8 KB (fp32, residual)
    __shared__ __bf16 xsb[16 * kHID];     //  4 KB (bf16, GEMM1 A)
    __shared__ __bf16 convs[16 * kHC];    // 16 KB (bf16, GEMM2 A)
    __shared__ float  ys[16 * kHID];      //  8 KB
    __shared__ float  mu[16], rs[16];

    int tid = threadIdx.x;
    int m0  = blockIdx.x * 16;
    for (int i = tid; i < 16 * kHID; i += 256) {
        float xv = x[(size_t)(m0 + (i >> 7)) * kHID + (i & 127)];
        xs[i]  = xv;
        xsb[i] = (__bf16)xv;
    }
    __syncthreads();

    int w    = tid >> 5;
    int lane = tid & 31;
    int col  = lane & 15;
    int mb   = (lane >> 4) << 3;

    // ---- GEMM 1: x @ Wskip, wave w owns 16x64 (4 ntiles) ----
    v8f acc[4] = {{}, {}, {}, {}};
#pragma unroll
    for (int ks = 0; ks < 4; ++ks) {         // K = 128
        v16bf a;
        load_a_frag(a, xsb, kHID, ks << 5, lane);
#pragma unroll
        for (int t = 0; t < 4; ++t) {
            v16bf b = load_b_packed(WskipP, (w * 4 + t) * 4 + ks, lane);
            acc[t] = wmma_bf16(a, b, acc[t]);
        }
    }
#pragma unroll
    for (int t = 0; t < 4; ++t) {
        int n = (w * 4 + t) * 16 + col;
        float bs = bskip[n];
#pragma unroll
        for (int r = 0; r < 8; ++r) {
            int m = mb + r;
            convs[m * kHC + n] =
                (__bf16)(acc[t][r] + agg[(size_t)(m0 + m) * kHC + n] + bs);
        }
    }
    __syncthreads();

    // ---- GEMM 2: conv @ Wproj, wave w owns 16x16 tile (n0 = w*16) ----
    v8f acc2 = {};
    int n0 = w * 16;
#pragma unroll
    for (int ks = 0; ks < 16; ++ks) {        // K = 512
        v16bf a, b;
        load_a_frag(a, convs, kHC, ks << 5, lane);
        b = load_b_packed(WprojP, w * 16 + ks, lane);
        acc2 = wmma_bf16(a, b, acc2);
    }
    {
        int n = n0 + col;
        float bp = bproj[n];
#pragma unroll
        for (int r = 0; r < 8; ++r) {
            int m = mb + r;
            float hv = acc2[r] + bp;
            hv = hv > 0.0f ? hv : (__expf(hv) - 1.0f);   // ELU(alpha=1)
            ys[m * kHID + n] = xs[m * kHID + n] + hv;    // residual
        }
    }
    __syncthreads();

    // ---- LayerNorm over 128 channels per row ----
    if (tid < 16) {
        float s = 0.0f, s2 = 0.0f;
        for (int c = 0; c < kHID; ++c) {
            float yv = ys[tid * kHID + c];
            s += yv; s2 += yv * yv;
        }
        float m_ = s * (1.0f / kHID);
        float var = s2 * (1.0f / kHID) - m_ * m_;
        mu[tid] = m_;
        rs[tid] = rsqrtf(var + 1e-5f);
    }
    __syncthreads();
    for (int i = tid; i < 16 * kHID; i += 256) {
        int r = i >> 7, c = i & 127;
        out[(size_t)(m0 + r) * kHID + c] =
            (ys[i] - mu[r]) * rs[r] * gamma[c] + beta[c];
    }
}

// ---------------- host launcher --------------------

extern "C" void kernel_launch(void* const* d_in, const int* in_sizes, int n_in,
                              void* d_out, int out_size, void* d_ws, size_t ws_size,
                              hipStream_t stream) {
    (void)in_sizes; (void)n_in; (void)out_size; (void)ws_size;

    const float* x0    = (const float*)d_in[0];
    const int*   ei    = (const int*)  d_in[1];
    const float* ea    = (const float*)d_in[2];
    const float* Wq    = (const float*)d_in[3];
    const float* bq    = (const float*)d_in[4];
    const float* Wk    = (const float*)d_in[5];
    const float* bk    = (const float*)d_in[6];
    const float* Wv    = (const float*)d_in[7];
    const float* bv    = (const float*)d_in[8];
    const float* We    = (const float*)d_in[9];
    const float* Wsk   = (const float*)d_in[10];
    const float* bsk   = (const float*)d_in[11];
    const float* Wpr   = (const float*)d_in[12];
    const float* bpr   = (const float*)d_in[13];
    const float* gamma = (const float*)d_in[14];
    const float* beta  = (const float*)d_in[15];
    float* outp = (float*)d_out;

    // workspace layout
    char* ws = (char*)d_ws;
    size_t off = 0;
    float*    q      = (float*)(ws + off); off += (size_t)kN * kHC * 4;
    float*    k      = (float*)(ws + off); off += (size_t)kN * kHC * 4;
    float*    v      = (float*)(ws + off); off += (size_t)kN * kHC * 4;
    float*    agg    = (float*)(ws + off); off += (size_t)kN * kHC * 4;
    unsigned* maxkey = (unsigned*)(ws + off); off += (size_t)kN * kH * 4;
    float*    denom  = (float*)(ws + off); off += (size_t)kN * kH * 4;
    float*    alpha  = (float*)(ws + off); off += (size_t)kE * kH * 4;
    float*    pbuf   = (float*)(ws + off); off += (size_t)kE * kH * 4;
    float*    xbuf   = (float*)(ws + off); off += (size_t)kN * kHID * 4;
    // packed bf16 weights (fragment order), 128 KB each
    __bf16* WqP = (__bf16*)(ws + off); off += (size_t)kHID * kHC * 2;
    __bf16* WkP = (__bf16*)(ws + off); off += (size_t)kHID * kHC * 2;
    __bf16* WvP = (__bf16*)(ws + off); off += (size_t)kHID * kHC * 2;
    __bf16* WsP = (__bf16*)(ws + off); off += (size_t)kHID * kHC * 2;
    __bf16* WpP = (__bf16*)(ws + off); off += (size_t)kHC * kHID * 2;

    const int nzero  = kN * kHC + kN * kH * 2;  // agg + maxkey + denom
    const int npackB = (kHID * kHC + 255) / 256;

    for (int l = 0; l < kL; ++l) {
        const float* xin  = (l == 0) ? x0 : xbuf;
        float*       xout = (l == kL - 1) ? outp : xbuf;

        const float* Wq_l = Wq  + (size_t)l * kHID * kHC;
        const float* Wk_l = Wk  + (size_t)l * kHID * kHC;
        const float* Wv_l = Wv  + (size_t)l * kHID * kHC;
        const float* bq_l = bq  + (size_t)l * kHC;
        const float* bk_l = bk  + (size_t)l * kHC;
        const float* bv_l = bv  + (size_t)l * kHC;
        const float* We_l = We  + (size_t)l * kHC;          // ED = 1
        const float* Ws_l = Wsk + (size_t)l * kHID * kHC;
        const float* bs_l = bsk + (size_t)l * kHC;
        const float* Wp_l = Wpr + (size_t)l * kHC * kHID;
        const float* bp_l = bpr + (size_t)l * kHID;
        const float* g_l  = gamma + (size_t)l * kHID;
        const float* be_l = beta  + (size_t)l * kHID;

        zero_kernel<<<(nzero + 255) / 256, 256, 0, stream>>>(agg, nzero);

        pack_w_kernel<<<npackB, 256, 0, stream>>>(Wq_l, kHID, kHC, WqP);
        pack_w_kernel<<<npackB, 256, 0, stream>>>(Wk_l, kHID, kHC, WkP);
        pack_w_kernel<<<npackB, 256, 0, stream>>>(Wv_l, kHID, kHC, WvP);
        pack_w_kernel<<<npackB, 256, 0, stream>>>(Ws_l, kHID, kHC, WsP);
        pack_w_kernel<<<npackB, 256, 0, stream>>>(Wp_l, kHC, kHID, WpP);

        qkv_gemm_kernel<<<dim3(kN / 16, 12), 256, 0, stream>>>(
            xin, (const v16bf*)WqP, (const v16bf*)WkP, (const v16bf*)WvP,
            bq_l, bk_l, bv_l, q, k, v);

        edge_alpha_kernel<<<kE / 8, 256, 0, stream>>>(
            ei, ea, q, k, We_l, alpha, maxkey);

        edge_exp_kernel<<<(kE * kH + 255) / 256, 256, 0, stream>>>(
            ei, alpha, maxkey, pbuf, denom);

        edge_scatter_kernel<<<kE / 8, 256, 0, stream>>>(
            ei, ea, v, We_l, pbuf, denom, agg);

        conv_proj_ln_kernel<<<kN / 16, 256, 0, stream>>>(
            xin, agg, (const v16bf*)WsP, bs_l, (const v16bf*)WpP, bp_l,
            g_l, be_l, xout);
    }
}
